// NystromAttention_57810259804642
// MI455X (gfx1250) — matrix operand order repaired and checked by reading, not compiled, not run on hardware
//
#include <hip/hip_runtime.h>

// ---------------------------------------------------------------------------
// Nystrom attention forward for MI455X (gfx1250, wave32, WMMA).
//   1. w_qkv / w_out  f32 -> bf16, TRANSPOSED to [n][k] so GEMM B-fragments
//      are K-contiguous (2x ds_load_b128 instead of 16x ds_load_u16)
//   2. qkv GEMM (bf16 WMMA, templated epilogue) -> q,k,v bf16, q pre-scaled
//   3. landmarks -> q_land, k_land bf16
//   4. sim2 + softmax -> attn2 f32
//   5. pinv: 6 Newton iterations on V_WMMA_F32_16X16X4_F32 (L2-resident)
//   6. flash-style fused softmax(q_land k^T) @ v -> r
//   7. W = z @ r -> bf16
//   8. depthwise 33-tap conv on v -> residual bf16
//   9. fused softmax(q k_land^T) @ W + residual -> [b,n,512] bf16
//  10. out-proj GEMM (bf16 WMMA) + bias -> d_out f32
// ---------------------------------------------------------------------------

typedef __attribute__((ext_vector_type(16))) __bf16  v16bf;
typedef __attribute__((ext_vector_type(8)))  __bf16  v8bf;
typedef __attribute__((ext_vector_type(4)))  __bf16  v4bf;
typedef __attribute__((ext_vector_type(8)))  float   v8f;
typedef __attribute__((ext_vector_type(4)))  float   v4f;
typedef __attribute__((ext_vector_type(2)))  float   v2f;

constexpr int CB   = 4;
constexpr int CN   = 8192;
constexpr int CDIM = 512;
constexpr int CH   = 8;
constexpr int CDH  = 64;
constexpr int CM   = 256;
constexpr int CL   = 32;     // N / M
constexpr int CKW  = 33;
constexpr int CIN  = 512;    // H * DH
constexpr int CTIN = 1536;   // 3 * INNER
constexpr int CBH  = CB * CH;
constexpr int LPAD = 40;     // LDS tile stride (bf16 elems): conflict-free b128

// ------------------------------ WMMA helpers -------------------------------

__device__ __forceinline__ v8f vzero8() {
  v8f z = {0.f,0.f,0.f,0.f,0.f,0.f,0.f,0.f};
  return z;
}

__device__ __forceinline__ v8f wmma_bf16(v16bf a, v16bf b, v8f c) {
  return __builtin_amdgcn_wmma_f32_16x16x32_bf16(false, a, false, b,
                                                 (short)0, c, false, false);
}

__device__ __forceinline__ v8f wmma_f32k4(v2f a, v2f b, v8f c) {
  return __builtin_amdgcn_wmma_f32_16x16x4_f32(false, a, false, b,
                                               (short)0, c, false, false);
}

// A-matrix fragment (16x32 bf16, row-major source with given stride).
// Layout: lanes 0-15 row M=lane, elems 0-15 = K0..15;
//         lanes 16-31 same rows, K16..31.
__device__ __forceinline__ v16bf frag_a_bf(const __bf16* base, int stride,
                                           int row, int lane) {
  const __bf16* p = base + (size_t)(row + (lane & 15)) * stride
                         + ((lane < 16) ? 0 : 16);
  v16bf a;
#pragma unroll
  for (int e = 0; e < 8; e++) { a[e] = p[e]; a[e + 8] = p[e + 8]; }
  return a;
}

// Same fragment but from an f32 source (converted to bf16 on the fly).
__device__ __forceinline__ v16bf frag_a_f32(const float* base, int stride,
                                            int row, int lane) {
  const float* p = base + (size_t)(row + (lane & 15)) * stride
                        + ((lane < 16) ? 0 : 16);
  v16bf a;
#pragma unroll
  for (int e = 0; e < 8; e++) {
    a[e]     = (__bf16)p[e];
    a[e + 8] = (__bf16)p[e + 8];
  }
  return a;
}

// B-matrix fragment (32x16) from a source where the WMMA K-dim is CONTIGUOUS
// and N is strided: B[k][n] = base[n*rstride + k].
// Order used here: lanes 0-15 col N=lane, elems = K0..15; lanes 16-31 K16..31.
__device__ __forceinline__ v16bf frag_bT_bf(const __bf16* base, int rstride,
                                            int col, int lane) {
  const __bf16* p = base + (size_t)(col + (lane & 15)) * rstride
                         + ((lane < 16) ? 0 : 16);
  v16bf b;
#pragma unroll
  for (int e = 0; e < 16; e++) b[e] = p[e];
  return b;
}

// B-matrix fragment (32x16) from K-major source: B[k][n] at k*stride + n.
__device__ __forceinline__ v16bf frag_b_bf(const __bf16* base, int stride,
                                           int col, int lane) {
  const __bf16* p = base + (size_t)((lane < 16) ? 0 : 16) * stride
                         + col + (lane & 15);
  v16bf b;
#pragma unroll
  for (int e = 0; e < 16; e++) b[e] = p[(size_t)e * stride];
  return b;
}

// ------------------------------ small kernels ------------------------------

__global__ void f32_to_bf16_kernel(const float* __restrict__ src,
                                   __bf16* __restrict__ dst, int n) {
  int i = blockIdx.x * 256 + threadIdx.x;
  if (i < n) dst[i] = (__bf16)src[i];
}

// src [K][Ncols] f32 -> dst [Ncols][K] bf16  (K-contiguous weights for GEMM B)
__global__ void convert_transpose_kernel(const float* __restrict__ src,
                                         __bf16* __restrict__ dst,
                                         int K, int Ncols) {
  int idx = blockIdx.x * 256 + threadIdx.x;
  if (idx >= K * Ncols) return;
  int k = idx / Ncols, n = idx % Ncols;
  dst[(size_t)n * K + k] = (__bf16)src[idx];
}

__global__ void init_smax_kernel(unsigned* s) {
  if (threadIdx.x < 2) s[threadIdx.x] = 0u;
}

// t = c*I - src   (batched 256x256)
__global__ void identsub_kernel(const float* __restrict__ src,
                                float* __restrict__ dst, float c) {
  int idx = blockIdx.x * 256 + threadIdx.x;
  int j = idx & 255;
  int i = (idx >> 8) & 255;
  dst[idx] = ((i == j) ? c : 0.f) - src[idx];
}

// z = attn2^T / (max_colsum * max_rowsum)
__global__ void zinit_kernel(const float* __restrict__ X,
                             float* __restrict__ z,
                             const unsigned* __restrict__ smax) {
  int idx = blockIdx.x * 256 + threadIdx.x;
  int j = idx & 255;
  int i = (idx >> 8) & 255;
  int bh = idx >> 16;
  float s = __uint_as_float(smax[0]) * __uint_as_float(smax[1]);
  z[idx] = X[((size_t)bh * CM + j) * CM + i] / s;
}

// global max of per-row and per-column abs sums of attn2 (pinv init scale)
__global__ __launch_bounds__(256)
void pinv_scale_kernel(const float* __restrict__ X, unsigned* smax) {
  int bh = blockIdx.x;
  int j = threadIdx.x;
  const float* Xp = X + (size_t)bh * CM * CM;
  float cs = 0.f, rs = 0.f;
  for (int i = 0; i < CM; i++) {
    cs += fabsf(Xp[(size_t)i * CM + j]);   // column sum  (ref: row_sum)
    rs += fabsf(Xp[(size_t)j * CM + i]);   // row sum     (ref: col_sum)
  }
  __shared__ float red[256];
  red[j] = cs; __syncthreads();
  for (int o = 128; o; o >>= 1) {
    if (j < o) red[j] = fmaxf(red[j], red[j + o]);
    __syncthreads();
  }
  if (j == 0) atomicMax(&smax[0], __float_as_uint(red[0]));
  __syncthreads();
  red[j] = rs; __syncthreads();
  for (int o = 128; o; o >>= 1) {
    if (j < o) red[j] = fmaxf(red[j], red[j + o]);
    __syncthreads();
  }
  if (j == 0) atomicMax(&smax[1], __float_as_uint(red[0]));
}

// landmarks: mean over CL consecutive tokens for q and k
__global__ void landmark_kernel(const __bf16* __restrict__ q,
                                const __bf16* __restrict__ k,
                                __bf16* __restrict__ ql,
                                __bf16* __restrict__ kl) {
  int bhm = blockIdx.x;
  int d = threadIdx.x;
  int m = bhm & (CM - 1);
  int bh = bhm >> 8;
  size_t base = ((size_t)bh * CN + (size_t)m * CL) * CDH + d;
  float sq = 0.f, sk = 0.f;
  for (int j = 0; j < CL; j++) {
    sq += (float)q[base + (size_t)j * CDH];
    sk += (float)k[base + (size_t)j * CDH];
  }
  size_t o = ((size_t)bh * CM + m) * CDH + d;
  ql[o] = (__bf16)(sq * (1.f / CL));
  kl[o] = (__bf16)(sk * (1.f / CL));
}

// attn2 = softmax(q_land @ k_land^T) -- one row per block
__global__ __launch_bounds__(256)
void sim2_softmax_kernel(const __bf16* __restrict__ ql,
                         const __bf16* __restrict__ kl,
                         float* __restrict__ attn2) {
  int bh = blockIdx.x >> 8;
  int i  = blockIdx.x & (CM - 1);
  int j  = threadIdx.x;
  const __bf16* qr = ql + ((size_t)bh * CM + i) * CDH;
  const __bf16* kr = kl + ((size_t)bh * CM + j) * CDH;
  float s = 0.f;
  for (int d = 0; d < CDH; d++) s += (float)qr[d] * (float)kr[d];
  __shared__ float red[256];
  red[j] = s; __syncthreads();
  for (int o = 128; o; o >>= 1) {
    if (j < o) red[j] = fmaxf(red[j], red[j + o]);
    __syncthreads();
  }
  float mx = red[0]; __syncthreads();
  float e = __expf(s - mx);
  red[j] = e; __syncthreads();
  for (int o = 128; o; o >>= 1) {
    if (j < o) red[j] = red[j] + red[j + o];
    __syncthreads();
  }
  attn2[((size_t)bh * CM + i) * CM + j] = e / red[0];
}

// depthwise residual conv over the sequence axis (kernel 33, per head)
__global__ __launch_bounds__(256)
void conv_res_kernel(const __bf16* __restrict__ v,
                     const float* __restrict__ w_res,
                     __bf16* __restrict__ resb) {
  int blk = blockIdx.x;                  // CBH * CN/4 blocks
  int bh = blk / (CN / 4);
  int nq = blk % (CN / 4);
  int d = threadIdx.x & 63;
  int n = nq * 4 + (threadIdx.x >> 6);
  int h = bh & (CH - 1);
  __shared__ float wsh[CKW];
  if (threadIdx.x < CKW) wsh[threadIdx.x] = w_res[h * CKW + threadIdx.x];
  __syncthreads();
  const __bf16* vp = v + (size_t)bh * CN * CDH;
  float acc = 0.f;
#pragma unroll
  for (int t = 0; t < CKW; t++) {
    int nn = n + t - CKW / 2;
    if (nn >= 0 && nn < CN) acc += wsh[t] * (float)vp[(size_t)nn * CDH + d];
  }
  resb[((size_t)bh * CN + n) * CDH + d] = (__bf16)acc;
}

// ----------------------- big bf16 WMMA GEMM (templated) --------------------
// C[Mr,Nc] = A[Mr,Kd] * B[Kd,Nc]; B given TRANSPOSED [Nc][Kd] (K-contiguous).
// 128x128 block tile, 8 waves, K-step 32, vectorized b128 staging.
// MODE 0: A = f32, scatter qkv into [b,h,n,d] bf16 (q scaled by 0.125)
// MODE 1: A = bf16, f32 out + bias
template <int MODE>
__global__ __launch_bounds__(256)
void gemm_bf16_kernel(const void* __restrict__ Aptr,
                      const __bf16* __restrict__ BT,
                      int Nc, int Kd,
                      __bf16* __restrict__ qb, __bf16* __restrict__ kb,
                      __bf16* __restrict__ vb,
                      float* __restrict__ outp, const float* __restrict__ bias) {
  __shared__ __bf16 As[128][LPAD];
  __shared__ __bf16 BsT[128][LPAD];
  const int tid  = threadIdx.x;
  const int lane = tid & 31;
  const int wave = tid >> 5;
  const int bm = blockIdx.y * 128;
  const int bn = blockIdx.x * 128;
  const int wr = (wave & 3) * 32;   // wave row offset
  const int wc = (wave >> 2) * 64;  // wave col offset

  v8f acc[2][4];
#pragma unroll
  for (int i = 0; i < 2; i++)
#pragma unroll
    for (int j = 0; j < 4; j++) acc[i][j] = vzero8();

  for (int k0 = 0; k0 < Kd; k0 += 32) {
    if (MODE == 0) {
      const float* Af = (const float*)Aptr;
#pragma unroll
      for (int i = 0; i < 4; i++) {          // 1024 float4 slots
        int slot = tid + i * 256;
        int r = slot >> 3;                   // 8 slots per 32-wide row
        int c = (slot & 7) * 4;
        v4f f = *(const v4f*)&Af[(size_t)(bm + r) * Kd + k0 + c];
        v4bf o;
        o[0] = (__bf16)f[0]; o[1] = (__bf16)f[1];
        o[2] = (__bf16)f[2]; o[3] = (__bf16)f[3];
        *(v4bf*)&As[r][c] = o;
      }
    } else {
      const __bf16* Ab = (const __bf16*)Aptr;
#pragma unroll
      for (int i = 0; i < 2; i++) {          // 512 v8bf slots
        int slot = tid + i * 256;
        int r = slot >> 2;                   // 4 slots per row
        int c = (slot & 3) * 8;
        *(v8bf*)&As[r][c] = *(const v8bf*)&Ab[(size_t)(bm + r) * Kd + k0 + c];
      }
    }
#pragma unroll
    for (int i = 0; i < 2; i++) {            // B tile: 128 n-rows x 32 k
      int slot = tid + i * 256;
      int r = slot >> 2;
      int c = (slot & 3) * 8;
      *(v8bf*)&BsT[r][c] = *(const v8bf*)&BT[(size_t)(bn + r) * Kd + k0 + c];
    }
    if (k0 + 32 < Kd) {                      // prefetch next B tile
      __builtin_prefetch(&BT[(size_t)(bn + (tid >> 1)) * Kd + k0 + 32], 0, 3);
    }
    __syncthreads();
    v16bf afr[2], bfr[4];
#pragma unroll
    for (int t = 0; t < 2; t++)
      afr[t] = frag_a_bf(&As[0][0], LPAD, wr + 16 * t, lane);
#pragma unroll
    for (int t = 0; t < 4; t++)
      bfr[t] = frag_bT_bf(&BsT[0][0], LPAD, wc + 16 * t, lane);
#pragma unroll
    for (int i = 0; i < 2; i++)
#pragma unroll
      for (int j = 0; j < 4; j++) acc[i][j] = wmma_bf16(afr[i], bfr[j], acc[i][j]);
    __syncthreads();
  }

  const int rb = (lane < 16) ? 0 : 8;
  const int cl = lane & 15;
#pragma unroll
  for (int i = 0; i < 2; i++) {
#pragma unroll
    for (int j = 0; j < 4; j++) {
#pragma unroll
      for (int e = 0; e < 8; e++) {
        int gr = bm + wr + 16 * i + rb + e;
        int gc = bn + wc + 16 * j + cl;
        float val = acc[i][j][e];
        if (MODE == 0) {
          int which = gc >> 9;               // 0=q 1=k 2=v
          int rem = gc & 511;
          int hh = rem >> 6, dd = rem & 63;
          int bb = gr >> 13, nn = gr & (CN - 1);
          __bf16* dst = (which == 0) ? qb : (which == 1) ? kb : vb;
          if (which == 0) val *= 0.125f;     // dh^-0.5
          dst[(((size_t)bb * CH + hh) * CN + nn) * CDH + dd] = (__bf16)val;
        } else {
          outp[(size_t)gr * Nc + gc] = val + bias[gc];
        }
      }
    }
  }
}

// ------------------- batched 256x256 f32 WMMA matmul -----------------------
// D[256,Nc] = alpha * A[256,256] @ B[256,Nc], batched over blockIdx.z.
// Pure-global: the pinv working set (~48 MB) is L2-resident (192 MB L2).
__global__ __launch_bounds__(256)
void mm256_f32_kernel(const float* __restrict__ A, const float* __restrict__ Bm,
                      float* __restrict__ D, int Nc, float alpha) {
  int bh = blockIdx.z;
  const float* Ap = A  + (size_t)bh * CM * CM;
  const float* Bp = Bm + (size_t)bh * CM * Nc;
  float*       Dp = D  + (size_t)bh * CM * Nc;
  int bm = blockIdx.y * 64, bn = blockIdx.x * 64;
  int lane = threadIdx.x & 31, wave = threadIdx.x >> 5;
  int ml = lane & 15;
  int t0 = wave * 2;
  v8f acc[2];
  acc[0] = vzero8(); acc[1] = vzero8();
  for (int kk = 0; kk < CM; kk += 4) {
    int ka = kk + ((lane < 16) ? 0 : 2);
#pragma unroll
    for (int t = 0; t < 2; t++) {
      int ti = t0 + t;
      int trow = (ti >> 2) * 16, tcol = (ti & 3) * 16;
      v2f a, b;
      int ar = bm + trow + ml;
      a[0] = Ap[(size_t)ar * CM + ka];
      a[1] = Ap[(size_t)ar * CM + ka + 1];
      int bc = bn + tcol + ml;
      b[0] = Bp[(size_t)ka * Nc + bc];
      b[1] = Bp[(size_t)(ka + 1) * Nc + bc];
      acc[t] = wmma_f32k4(a, b, acc[t]);
    }
  }
  const int rb = (lane < 16) ? 0 : 8;
#pragma unroll
  for (int t = 0; t < 2; t++) {
    int ti = t0 + t;
    int trow = (ti >> 2) * 16, tcol = (ti & 3) * 16;
#pragma unroll
    for (int e = 0; e < 8; e++)
      Dp[(size_t)(bm + trow + rb + e) * Nc + bn + tcol + ml] = alpha * acc[t][e];
  }
}

// -------- flash-style fused r = softmax(q_land @ k^T) @ v  ------------------
// One block = 16 landmark rows of one (b,h); online softmax over n in 64-chunks.
__global__ __launch_bounds__(128)
void flash_a3v_kernel(const __bf16* __restrict__ ql,
                      const __bf16* __restrict__ k,
                      const __bf16* __restrict__ v,
                      float* __restrict__ r_out) {
  int bh = blockIdx.y;
  int m0 = blockIdx.x * 16;
  int lane = threadIdx.x & 31, wave = threadIdx.x >> 5;  // 4 waves

  const __bf16* Q  = ql + ((size_t)bh * CM + m0) * CDH;
  const __bf16* Kp = k  + (size_t)bh * CN * CDH;
  const __bf16* Vp = v  + (size_t)bh * CN * CDH;

  v16bf qf[2];
#pragma unroll
  for (int s = 0; s < 2; s++) qf[s] = frag_a_bf(Q + 32 * s, CDH, 0, lane);

  __shared__ float  Sld[16][64];
  __shared__ __bf16 Pld[16][64];
  __shared__ float  rowscale[16];
  __shared__ float  rowm[16], rowl[16];
  if (threadIdx.x < 16) { rowm[threadIdx.x] = -1e30f; rowl[threadIdx.x] = 0.f; }
  v8f o = vzero8();     // wave's output tile: 16 rows x d = 16*wave..+15
  __syncthreads();

  for (int n0 = 0; n0 < CN; n0 += 64) {
    // S tile: cols n0 + 16*wave .. +15  (B = K^T, K-dim = d contiguous)
    v8f s = vzero8();
#pragma unroll
    for (int ks = 0; ks < 2; ks++) {
      v16bf bf = frag_bT_bf(Kp + (size_t)n0 * CDH + 32 * ks, CDH, 16 * wave, lane);
      s = wmma_bf16(qf[ks], bf, s);
    }
    {
      int rbase = (lane < 16) ? 0 : 8;
      int cc = 16 * wave + (lane & 15);
#pragma unroll
      for (int e = 0; e < 8; e++) Sld[rbase + e][cc] = s[e];
    }
    __syncthreads();

    if (threadIdx.x < 16) {       // online-softmax row update
      int rr = threadIdx.x;
      float mold = rowm[rr];
      float mx = mold;
      for (int c = 0; c < 64; c++) mx = fmaxf(mx, Sld[rr][c]);
      float sc = __expf(mold - mx);
      float ls = rowl[rr] * sc;
      for (int c = 0; c < 64; c++) {
        float e = __expf(Sld[rr][c] - mx);
        Pld[rr][c] = (__bf16)e;
        ls += e;
      }
      rowm[rr] = mx; rowl[rr] = ls; rowscale[rr] = sc;
    }
    __syncthreads();

    {
      int rbase = (lane < 16) ? 0 : 8;
#pragma unroll
      for (int e = 0; e < 8; e++) o[e] *= rowscale[rbase + e];
    }
#pragma unroll
    for (int ks = 0; ks < 2; ks++) {      // o += P @ Vchunk
      v16bf af = frag_a_bf(&Pld[0][32 * ks], 64, 0, lane);
      v16bf bf = frag_b_bf(Vp + (size_t)(n0 + 32 * ks) * CDH, CDH, 16 * wave, lane);
      o = wmma_bf16(af, bf, o);
    }
    __syncthreads();
  }

  {
    int rbase = (lane < 16) ? 0 : 8;
    int dd = 16 * wave + (lane & 15);
    float* rp = r_out + ((size_t)bh * CM + m0) * CDH;
#pragma unroll
    for (int e = 0; e < 8; e++)
      rp[(size_t)(rbase + e) * CDH + dd] = o[e] / rowl[rbase + e];
  }
}

// ---- fused out = softmax(q @ k_land^T) @ W + conv_residual  -> bf16 --------
// One block = 32 sequence rows of one (b,h). S kept in LDS; attn1 never
// touches HBM. Output scattered into [b, n, h*64+d] for the final GEMM.
__global__ __launch_bounds__(256)
void attn1_fused_kernel(const __bf16* __restrict__ qb,
                        const __bf16* __restrict__ klb,
                        const __bf16* __restrict__ wbf,
                        const __bf16* __restrict__ resb,
                        __bf16* __restrict__ outin) {
  int bh = blockIdx.y;
  int h = bh & (CH - 1);
  int b = bh >> 3;
  int n0 = blockIdx.x * 32;
  int lane = threadIdx.x & 31, wave = threadIdx.x >> 5;   // 8 waves

  __shared__ float Sld[32][260];
  __shared__ float invl[32];

  const __bf16* Qp = qb  + ((size_t)bh * CN + n0) * CDH;
  const __bf16* KL = klb + (size_t)bh * CM * CDH;

  // Phase A: S[32][256] = Q @ KL^T ; wave covers col tiles {2w, 2w+1}
#pragma unroll
  for (int tr = 0; tr < 2; tr++) {
#pragma unroll
    for (int tci = 0; tci < 2; tci++) {
      int tc = 2 * wave + tci;
      v8f s = vzero8();
#pragma unroll
      for (int ks = 0; ks < 2; ks++) {
        v16bf a  = frag_a_bf(Qp + 32 * ks, CDH, tr * 16, lane);
        v16bf bf = frag_bT_bf(KL + 32 * ks, CDH, tc * 16, lane);
        s = wmma_bf16(a, bf, s);
      }
      int rb = tr * 16 + ((lane < 16) ? 0 : 8);
      int cc = tc * 16 + (lane & 15);
#pragma unroll
      for (int e = 0; e < 8; e++) Sld[rb + e][cc] = s[e];
    }
  }
  __syncthreads();

  // Phase B: softmax rows, exp written back into Sld (as P, f32)
  if (threadIdx.x < 32) {
    int rr = threadIdx.x;
    float mx = -1e30f;
    for (int c = 0; c < CM; c++) mx = fmaxf(mx, Sld[rr][c]);
    float sum = 0.f;
    for (int c = 0; c < CM; c++) {
      float e = __expf(Sld[rr][c] - mx);
      Sld[rr][c] = e;
      sum += e;
    }
    invl[rr] = 1.f / sum;
  }
  __syncthreads();

  // Phase C: O[32][64] = P @ W ; wave -> (tr = wave>>2, tc = wave&3)
  int tr = wave >> 2, tc = wave & 3;
  v8f o = vzero8();
  const __bf16* Wp = wbf + (size_t)bh * CM * CDH;
#pragma unroll
  for (int ks = 0; ks < 8; ks++) {
    v16bf a  = frag_a_f32(&Sld[tr * 16][32 * ks], 260, 0, lane);
    v16bf bf = frag_b_bf(Wp + (size_t)(32 * ks) * CDH, CDH, tc * 16, lane);
    o = wmma_bf16(a, bf, o);
  }

  {
    int rb = tr * 16 + ((lane < 16) ? 0 : 8);
    int dd = tc * 16 + (lane & 15);
    const __bf16* rp = resb + ((size_t)bh * CN + n0) * CDH;
    __bf16* op = outin + ((size_t)b * CN + n0) * CIN + h * CDH;
#pragma unroll
    for (int e = 0; e < 8; e++) {
      int rr = rb + e;
      float val = o[e] * invl[rr] + (float)rp[(size_t)rr * CDH + dd];
      op[(size_t)rr * CIN + dd] = (__bf16)val;
    }
  }
}

// ------------------------------- host side ---------------------------------

extern "C" void kernel_launch(void* const* d_in, const int* in_sizes, int n_in,
                              void* d_out, int out_size, void* d_ws, size_t ws_size,
                              hipStream_t stream) {
  (void)in_sizes; (void)n_in; (void)out_size; (void)ws_size;
  const float* x      = (const float*)d_in[0];
  const float* w_qkv  = (const float*)d_in[1];
  const float* w_out  = (const float*)d_in[2];
  const float* b_out  = (const float*)d_in[3];
  const float* w_res  = (const float*)d_in[4];
  float* out = (float*)d_out;

  char* ws = (char*)d_ws;
  size_t off = 0;
  auto alloc = [&](size_t bytes) -> char* {
    off = (off + 255) & ~(size_t)255;
    char* p = ws + off;
    off += bytes;
    return p;
  };

  __bf16* wqkvT = (__bf16*)alloc((size_t)CDIM * CTIN * 2);  // [1536][512]
  __bf16* woutT = (__bf16*)alloc((size_t)CIN * CDIM * 2);   // [512][512]
  __bf16* qb    = (__bf16*)alloc((size_t)CBH * CN * CDH * 2);
  __bf16* kb    = (__bf16*)alloc((size_t)CBH * CN * CDH * 2);
  __bf16* vb    = (__bf16*)alloc((size_t)CBH * CN * CDH * 2);
  __bf16* qlb   = (__bf16*)alloc((size_t)CBH * CM * CDH * 2);
  __bf16* klb   = (__bf16*)alloc((size_t)CBH * CM * CDH * 2);
  float*  attn2 = (float*)alloc((size_t)CBH * CM * CM * 4);
  float*  zbuf  = (float*)alloc((size_t)CBH * CM * CM * 4);
  float*  xzbuf = (float*)alloc((size_t)CBH * CM * CM * 4);
  float*  ybuf  = (float*)alloc((size_t)CBH * CM * CM * 4);
  float*  tbuf  = (float*)alloc((size_t)CBH * CM * CM * 4);
  float*  ztbuf = (float*)alloc((size_t)CBH * CM * CM * 4);
  unsigned* smax = (unsigned*)alloc(256);
  float*  rbuf  = (float*)alloc((size_t)CBH * CM * CDH * 4);
  float*  wfbuf = (float*)alloc((size_t)CBH * CM * CDH * 4);
  __bf16* wbf   = (__bf16*)alloc((size_t)CBH * CM * CDH * 2);
  __bf16* resb  = (__bf16*)alloc((size_t)CBH * CN * CDH * 2);
  __bf16* outin = (__bf16*)alloc((size_t)CB * CN * CIN * 2);

  // 1. weight conversions (transposed to K-contiguous)
  convert_transpose_kernel<<<(CDIM * CTIN + 255) / 256, 256, 0, stream>>>(
      w_qkv, wqkvT, CDIM, CTIN);
  convert_transpose_kernel<<<(CIN * CDIM + 255) / 256, 256, 0, stream>>>(
      w_out, woutT, CIN, CDIM);

  // 2. qkv GEMM -> q,k,v (bf16, head-major, q scaled)
  gemm_bf16_kernel<0><<<dim3(CTIN / 128, (CB * CN) / 128), 256, 0, stream>>>(
      x, wqkvT, CTIN, CDIM, qb, kb, vb, nullptr, nullptr);

  // 3. landmarks
  landmark_kernel<<<CBH * CM, CDH, 0, stream>>>(qb, kb, qlb, klb);

  // 4. sim2 + softmax
  sim2_softmax_kernel<<<CBH * CM, CM, 0, stream>>>(qlb, klb, attn2);

  // 5. pinv: scales, z init, 6 Newton iterations on the f32 WMMA path
  init_smax_kernel<<<1, 32, 0, stream>>>(smax);
  pinv_scale_kernel<<<CBH, CM, 0, stream>>>(attn2, smax);
  zinit_kernel<<<(CBH * CM * CM) / 256, 256, 0, stream>>>(attn2, zbuf, smax);

  float* zcur = zbuf;
  float* zalt = ztbuf;
  const dim3 mmGrid(4, 4, CBH);
  const int isGrid = (CBH * CM * CM) / 256;
  for (int it = 0; it < 6; it++) {
    mm256_f32_kernel<<<mmGrid, 256, 0, stream>>>(attn2, zcur, xzbuf, CM, 1.f);
    identsub_kernel<<<isGrid, 256, 0, stream>>>(xzbuf, ybuf, 7.f);
    mm256_f32_kernel<<<mmGrid, 256, 0, stream>>>(xzbuf, ybuf, tbuf, CM, 1.f);
    identsub_kernel<<<isGrid, 256, 0, stream>>>(tbuf, ybuf, 15.f);
    mm256_f32_kernel<<<mmGrid, 256, 0, stream>>>(xzbuf, ybuf, tbuf, CM, 1.f);
    identsub_kernel<<<isGrid, 256, 0, stream>>>(tbuf, ybuf, 13.f);
    mm256_f32_kernel<<<mmGrid, 256, 0, stream>>>(zcur, ybuf, zalt, CM, 0.25f);
    float* tmp = zcur; zcur = zalt; zalt = tmp;
  }

  // 6. flash-style fused attn3 @ v -> r
  flash_a3v_kernel<<<dim3(CM / 16, CBH), 128, 0, stream>>>(qlb, kb, vb, rbuf);

  // 7. W = pinv(attn2) @ r ; -> bf16
  mm256_f32_kernel<<<dim3(1, 4, CBH), 256, 0, stream>>>(zcur, rbuf, wfbuf, CDH, 1.f);
  f32_to_bf16_kernel<<<(CBH * CM * CDH + 255) / 256, 256, 0, stream>>>(
      wfbuf, wbf, CBH * CM * CDH);

  // 8. depthwise residual conv on v
  conv_res_kernel<<<(CBH * CN) / 4, 256, 0, stream>>>(vb, w_res, resb);

  // 9. fused attn1 softmax @ W + residual -> [b, n, inner] bf16
  attn1_fused_kernel<<<dim3(CN / 32, CBH), 256, 0, stream>>>(qb, klb, wbf, resb, outin);

  // 10. output projection + bias -> d_out f32
  gemm_bf16_kernel<1><<<dim3(CDIM / 128, (CB * CN) / 128), 256, 0, stream>>>(
      outin, woutT, CDIM, CIN, nullptr, nullptr, nullptr, out, b_out);
}
